// MiniMaceEmbedding_37383395344578
// MI455X (gfx1250) — compile-verified, matrix operand order checked by reading, not run on hardware
//
#include <hip/hip_runtime.h>
#include <hip/hip_bf16.h>

#define NN_DIM 128
#define MSGC 8
#define NB 8
#define DCH 64
#define NCH 16
#define IRR 9
#define LATIN 224
#define MAXTP 2048
#define ECHUNK 8

typedef __attribute__((ext_vector_type(16))) _Float16 v16h;
typedef __attribute__((ext_vector_type(8)))  _Float16 v8h;
typedef __attribute__((ext_vector_type(8)))  float    v8f;

static __device__ __forceinline__ int imin_(int a, int b) { return a < b ? a : b; }
static __device__ __forceinline__ int imax_(int a, int b) { return a > b ? a : b; }

// ---------------------------------------------------------------------------
// Edge features: radial bessel basis * switch, spherical harmonics of unit vec
// ---------------------------------------------------------------------------
__global__ void edge_feat_kernel(const float* __restrict__ vec,
                                 const float* __restrict__ dist,
                                 const float* __restrict__ sw,
                                 float* __restrict__ rb,
                                 float* __restrict__ Y, int E) {
  int e = blockIdx.x * blockDim.x + threadIdx.x;
  if (e >= E) return;
  float d = dist[e];
  float s = sw[e];
  float inv = 1.f / d;
  float x = vec[3 * e + 0] * inv;
  float y = vec[3 * e + 1] * inv;
  float z = vec[3 * e + 2] * inv;
  const float PI5 = 3.14159265358979323846f / 5.0f;
  const float SQ = 0.63245553203367586640f;  // sqrt(2/5)
#pragma unroll
  for (int j = 0; j < NB; ++j)
    rb[(size_t)e * NB + j] = SQ * sinf((float)(j + 1) * PI5 * d) * inv * s;
  const float c1 = 1.73205080756887729353f;          // sqrt(3)
  const float c2 = 3.87298334620741688518f;          // sqrt(15)
  const float c3 = 1.11803398874989484820f;          // sqrt(5)/2
  const float c4 = 1.93649167310370844259f;          // sqrt(15)/2
  float* yy = Y + (size_t)e * IRR;
  yy[0] = 1.f;
  yy[1] = c1 * y;
  yy[2] = c1 * z;
  yy[3] = c1 * x;
  yy[4] = c2 * x * y;
  yy[5] = c2 * y * z;
  yy[6] = c3 * (3.f * z * z - 1.f);
  yy[7] = c2 * x * z;
  yy[8] = c4 * (x * x - y * y);
}

// ---------------------------------------------------------------------------
// CSR row offsets from sorted edge_src (binary search per node)
// ---------------------------------------------------------------------------
__global__ void row_offsets_kernel(const int* __restrict__ esrc, int* __restrict__ off,
                                   int Nn, int E) {
  int n = blockIdx.x * blockDim.x + threadIdx.x;
  if (n > Nn) return;
  if (n == Nn) { off[n] = E; return; }
  int lo = 0, hi = E;
  while (lo < hi) {
    int mid = (lo + hi) >> 1;
    if (esrc[mid] < n) lo = mid + 1; else hi = mid;
  }
  off[n] = lo;
}

// ---------------------------------------------------------------------------
// Species embedding: xi = W_spe[species] + b_spe
// ---------------------------------------------------------------------------
__global__ void embed_kernel(const int* __restrict__ species,
                             const float* __restrict__ Wspe,
                             const float* __restrict__ bspe,
                             float* __restrict__ xi, int Nn) {
  int idx = blockIdx.x * blockDim.x + threadIdx.x;
  if (idx >= Nn * NN_DIM) return;
  int n = idx >> 7, d = idx & 127;
  xi[idx] = Wspe[species[n] * NN_DIM + d] + bspe[d];
}

// ---------------------------------------------------------------------------
// f32 -> f16 activation pack (contiguous copy)
// ---------------------------------------------------------------------------
__global__ void pack_a_kernel(const float* __restrict__ A, _Float16* __restrict__ A16,
                              int total) {
  int idx = blockIdx.x * blockDim.x + threadIdx.x;
  if (idx < total) A16[idx] = (_Float16)A[idx];
}

// ---------------------------------------------------------------------------
// Weight pack: W[KxN] f32 -> Wt[N16xK] f16 transposed, zero-padded cols
// ---------------------------------------------------------------------------
__global__ void pack_w_kernel(const float* __restrict__ W, _Float16* __restrict__ Wt,
                              int K, int N, int N16) {
  int idx = blockIdx.x * blockDim.x + threadIdx.x;
  if (idx >= N16 * K) return;
  int n = idx / K, k = idx - n * K;
  Wt[idx] = (n < N) ? (_Float16)W[(size_t)k * N + n] : (_Float16)0.f;
}

// ---------------------------------------------------------------------------
// WMMA GEMM: out[MxN] = act(A[MxK] @ W[KxN] + bias[N]) (optionally += )
// A16: f16 row-major MxK. Wt: f16 N16xK (transposed, padded).
// One 16x16 C tile per wave; per K-step: 4x global_load_b128 + 1 wmma.
// ---------------------------------------------------------------------------
__global__ void __launch_bounds__(256)
gemm_wmma_kernel(const _Float16* __restrict__ A16, const _Float16* __restrict__ Wt,
                 const float* __restrict__ bias, float* __restrict__ out,
                 int M, int K, int N, int Ntiles, int tiles, int act, int accum) {
  int wave = (int)((blockIdx.x * blockDim.x + threadIdx.x) >> 5);
  if (wave >= tiles) return;               // wave-uniform: EXEC stays all-ones
  int lane = threadIdx.x & 31;
  int mt = wave / Ntiles;
  int nt = wave - mt * Ntiles;
  int half = lane >> 4;                    // 0 or 1
  int sub = lane & 15;
  int arow = mt * 16 + sub;
  if (arow >= M) arow = M - 1;             // clamp (M is a multiple of 16 here)
  int bcol = nt * 16 + sub;                // always valid: Wt padded to N16
  const _Float16* ap = A16 + (size_t)arow * K + half * 8;
  const _Float16* bp = Wt + (size_t)bcol * K + half * 8;
  v8f acc = {0.f, 0.f, 0.f, 0.f, 0.f, 0.f, 0.f, 0.f};
  for (int k0 = 0; k0 < K; k0 += 32) {
    v8h a0 = *(const v8h*)(ap + k0);
    v8h a1 = *(const v8h*)(ap + k0 + 16);
    v8h b0 = *(const v8h*)(bp + k0);
    v8h b1 = *(const v8h*)(bp + k0 + 16);
    v16h af = __builtin_shufflevector(a0, a1, 0, 1, 2, 3, 4, 5, 6, 7,
                                      8, 9, 10, 11, 12, 13, 14, 15);
    v16h bf = __builtin_shufflevector(b0, b1, 0, 1, 2, 3, 4, 5, 6, 7,
                                      8, 9, 10, 11, 12, 13, 14, 15);
    acc = __builtin_amdgcn_wmma_f32_16x16x32_f16(false, af, false, bf,
                                                 (short)0, acc, false, false);
  }
  if (bcol >= N) return;                   // only the padded-N store is masked
  float bv = bias[bcol];
#pragma unroll
  for (int r = 0; r < 8; ++r) {
    int orow = mt * 16 + r + half * 8;
    if (orow >= M) continue;
    float v = acc[r] + bv;
    if (act) v = v / (1.f + expf(-v));     // SiLU
    size_t o = (size_t)orow * N + bcol;
    if (accum) out[o] += v; else out[o] = v;
  }
}

// ---------------------------------------------------------------------------
// Layer-0 density: density[n,c,k] = sum_{e in row n} mi[dst][c/8]*rb[e][c%8]*Y[e][k]
// One block per node; per-edge operands staged in LDS; no atomics.
// ---------------------------------------------------------------------------
__global__ void density0_kernel(const int* __restrict__ roff, const int* __restrict__ edst,
                                const float* __restrict__ mi, const float* __restrict__ rb,
                                const float* __restrict__ Y, float* __restrict__ density) {
  __shared__ float smi[ECHUNK][MSGC];
  __shared__ float srb[ECHUNK][NB];
  __shared__ float sY[ECHUNK][IRR];
  int n = blockIdx.x;
  int t = threadIdx.x;  // 256
  int e0 = roff[n], e1 = roff[n + 1];
  int i0 = t, i1 = t + 256, i2 = t + 512;
  int c0 = i0 / IRR, k0 = i0 - c0 * IRR;
  int c1 = i1 / IRR, k1 = i1 - c1 * IRR;
  float a0 = 0.f, a1 = 0.f, a2 = 0.f;
  for (int eb = e0; eb < e1; eb += ECHUNK) {
    int cnt = imin_(ECHUNK, e1 - eb);
    __syncthreads();
    if (t < cnt * 25) {
      int j = t / 25, q = t - j * 25;
      int e = eb + j;
      if (q < 8)       { int d = edst[e]; smi[j][q] = mi[d * MSGC + q]; }
      else if (q < 16) { srb[j][q - 8]  = rb[(size_t)e * NB + (q - 8)]; }
      else             { sY[j][q - 16]  = Y[(size_t)e * IRR + (q - 16)]; }
    }
    __syncthreads();
    for (int j = 0; j < cnt; ++j) {
      a0 += smi[j][c0 >> 3] * srb[j][c0 & 7] * sY[j][k0];
      a1 += smi[j][c1 >> 3] * srb[j][c1 & 7] * sY[j][k1];
      if (i2 < DCH * IRR) {
        int c2 = i2 / IRR, k2 = i2 - c2 * IRR;
        a2 += smi[j][c2 >> 3] * srb[j][c2 & 7] * sY[j][k2];
      }
    }
  }
  size_t base = (size_t)n * (DCH * IRR);
  density[base + i0] = a0;
  density[base + i1] = a1;
  if (i2 < DCH * IRR) density[base + i2] = a2;
}

// ---------------------------------------------------------------------------
// Layer-1 density: density[n,c,k] += sum_e mi[dst][c/8]*rb[e][c%8]*rhoi[dst][c,k]
// rhoi gather is a coalesced 2.3KB block read per edge.
// ---------------------------------------------------------------------------
__global__ void density1_kernel(const int* __restrict__ roff, const int* __restrict__ edst,
                                const float* __restrict__ mi, const float* __restrict__ rb,
                                const float* __restrict__ rhoi, float* __restrict__ density) {
  __shared__ float smi[ECHUNK][MSGC];
  __shared__ float srb[ECHUNK][NB];
  __shared__ int sd[ECHUNK];
  int n = blockIdx.x;
  int t = threadIdx.x;  // 256
  int e0 = roff[n], e1 = roff[n + 1];
  int i0 = t, i1 = t + 256, i2 = t + 512;
  int c0 = i0 / IRR, c1 = i1 / IRR;
  float a0 = 0.f, a1 = 0.f, a2 = 0.f;
  for (int eb = e0; eb < e1; eb += ECHUNK) {
    int cnt = imin_(ECHUNK, e1 - eb);
    __syncthreads();
    if (t < cnt * 16) {
      int j = t >> 4, q = t & 15;
      int e = eb + j;
      if (q < 8) {
        int d = edst[e];
        smi[j][q] = mi[d * MSGC + q];
        if (q == 0) sd[j] = d;
      } else {
        srb[j][q - 8] = rb[(size_t)e * NB + (q - 8)];
      }
    }
    __syncthreads();
    for (int j = 0; j < cnt; ++j) {
      const float* rr = rhoi + (size_t)sd[j] * (DCH * IRR);
      a0 += smi[j][c0 >> 3] * srb[j][c0 & 7] * rr[i0];
      a1 += smi[j][c1 >> 3] * srb[j][c1 & 7] * rr[i1];
      if (i2 < DCH * IRR) {
        int c2 = i2 / IRR;
        a2 += smi[j][c2 >> 3] * srb[j][c2 & 7] * rr[i2];
      }
    }
  }
  size_t base = (size_t)n * (DCH * IRR);
  density[base + i0] += a0;
  density[base + i1] += a1;
  if (i2 < DCH * IRR) density[base + i2] += a2;
}

// ---------------------------------------------------------------------------
// Channel mixing: out[n,o,i] = sum_c x[n,c,i] * W[l(i) or 0][o][c]
// ---------------------------------------------------------------------------
__global__ void mix_kernel(const float* __restrict__ x, const float* __restrict__ W,
                           float* __restrict__ out, int Nn, int Cin, int Cout, int ldep) {
  int idx = blockIdx.x * blockDim.x + threadIdx.x;
  int total = Nn * Cout * IRR;
  if (idx >= total) return;
  int i = idx % IRR;
  int o = (idx / IRR) % Cout;
  int n = idx / (IRR * Cout);
  int l = ldep ? ((i == 0) ? 0 : ((i < 4) ? 1 : 2)) : 0;
  const float* w = W + ((size_t)l * Cout + o) * Cin;
  const float* xx = x + (size_t)n * Cin * IRR + i;
  float s = 0.f;
  for (int c = 0; c < Cin; ++c) s += xx[(size_t)c * IRR] * w[c];
  out[idx] = s;
}

// ---------------------------------------------------------------------------
// Real Clebsch-Gordan table generation on device (f64), once per launch.
// ---------------------------------------------------------------------------
static __device__ double dfact_(int v) {
  double r = 1.0;
  for (int i = 2; i <= v; ++i) r *= (double)i;
  return r;
}

static __device__ double cg1_dev(int l1, int m1, int l2, int m2, int l3, int m3) {
  if (m3 != m1 + m2) return 0.0;
  double pre = (double)(2 * l3 + 1) * dfact_(l1 + l2 - l3) * dfact_(l1 - l2 + l3) *
               dfact_(-l1 + l2 + l3) / dfact_(l1 + l2 + l3 + 1);
  pre *= dfact_(l3 + m3) * dfact_(l3 - m3) * dfact_(l1 - m1) * dfact_(l1 + m1) *
         dfact_(l2 - m2) * dfact_(l2 + m2);
  int kmin = imax_(0, imax_(l2 - l3 - m1, l1 - l3 + m2));
  int kmax = imin_(l1 + l2 - l3, imin_(l1 - m1, l2 + m2));
  double s = 0.0;
  for (int k = kmin; k <= kmax; ++k) {
    double den = dfact_(k) * dfact_(l1 + l2 - l3 - k) * dfact_(l1 - m1 - k) *
                 dfact_(l2 + m2 - k) * dfact_(l3 - l2 + m1 + k) * dfact_(l3 - l1 - m2 + k);
    s += ((k & 1) ? -1.0 : 1.0) / den;
  }
  return sqrt(pre) * s;
}

static __device__ void qmat_dev(int l, double Qr[5][5], double Qi[5][5]) {
  for (int a = 0; a < 5; ++a)
    for (int b = 0; b < 5; ++b) { Qr[a][b] = 0.0; Qi[a][b] = 0.0; }
  const double s2 = 0.70710678118654752440;
  Qr[l][l] = 1.0;
  for (int m = 1; m <= l; ++m) {
    double sg = (m & 1) ? -1.0 : 1.0;
    Qr[l + m][l + m] = sg * s2;
    Qr[l + m][l - m] = s2;
    Qi[l - m][l - m] = s2;
    Qi[l - m][l + m] = -sg * s2;
  }
}

__global__ void tp_init_kernel(int* __restrict__ cnt, int4* __restrict__ tidx,
                               float* __restrict__ tcf) {
  if (threadIdx.x != 0 || blockIdx.x != 0) return;
  int nnz = 0, p = 0;
  for (int l3 = 0; l3 <= 2; ++l3)
    for (int l1 = 0; l1 <= 2; ++l1)
      for (int l2 = 0; l2 <= 2; ++l2) {
        int dl = l1 - l2; if (dl < 0) dl = -dl;
        if (!(dl <= l3 && l3 <= l1 + l2)) continue;
        double C[5][5][5];
        for (int a = 0; a < 5; ++a)
          for (int b = 0; b < 5; ++b)
            for (int c = 0; c < 5; ++c) C[a][b][c] = 0.0;
        for (int m1 = -l1; m1 <= l1; ++m1)
          for (int m2 = -l2; m2 <= l2; ++m2) {
            int m3 = m1 + m2;
            if (m3 >= -l3 && m3 <= l3)
              C[m1 + l1][m2 + l2][m3 + l3] = cg1_dev(l1, m1, l2, m2, l3, m3);
          }
        double Q1r[5][5], Q1i[5][5], Q2r[5][5], Q2i[5][5], Q3r[5][5], Q3i[5][5];
        qmat_dev(l1, Q1r, Q1i);
        qmat_dev(l2, Q2r, Q2i);
        qmat_dev(l3, Q3r, Q3i);
        int par = (l1 + l2 + l3) & 1;
        for (int a = 0; a <= 2 * l1; ++a)
          for (int b = 0; b <= 2 * l2; ++b)
            for (int c = 0; c <= 2 * l3; ++c) {
              double re = 0.0, im = 0.0;
              for (int m = 0; m <= 2 * l1; ++m)
                for (int nn = 0; nn <= 2 * l2; ++nn)
                  for (int k = 0; k <= 2 * l3; ++k) {
                    double cg = C[m][nn][k];
                    if (cg == 0.0) continue;
                    double t1r = Q1r[a][m] * Q2r[b][nn] - Q1i[a][m] * Q2i[b][nn];
                    double t1i = Q1r[a][m] * Q2i[b][nn] + Q1i[a][m] * Q2r[b][nn];
                    re += (t1r * Q3r[c][k] + t1i * Q3i[c][k]) * cg;  // * conj(Q3)
                    im += (t1i * Q3r[c][k] - t1r * Q3i[c][k]) * cg;
                  }
              double v = par ? im : re;
              if (fabs(v) > 1e-12 && nnz < MAXTP) {
                tidx[nnz] = make_int4(p, l1 * l1 + a, l2 * l2 + b, l3 * l3 + c);
                tcf[nnz] = (float)v;
                ++nnz;
              }
            }
        ++p;
      }
  *cnt = nnz;
}

// ---------------------------------------------------------------------------
// Filtered tensor product: Li = TP(Vi,Hi); scal = Li[...,0]; Vi += Li
// ---------------------------------------------------------------------------
__global__ void tp_kernel(const float* __restrict__ Hi, const float* __restrict__ wtp,
                          const int* __restrict__ cnt, const int4* __restrict__ tidx,
                          const float* __restrict__ tcf, float* __restrict__ Vi,
                          float* __restrict__ scal, int Nn) {
  int idx = blockIdx.x * blockDim.x + threadIdx.x;
  if (idx >= Nn * NCH) return;
  float V[IRR], H[IRR], L[IRR];
  size_t base = (size_t)idx * IRR;
#pragma unroll
  for (int i = 0; i < IRR; ++i) {
    V[i] = Vi[base + i];
    H[i] = Hi[base + i];
    L[i] = 0.f;
  }
  int m = *cnt;
  for (int e = 0; e < m; ++e) {
    int4 tt = tidx[e];
    L[tt.w] += wtp[tt.x] * tcf[e] * V[tt.y] * H[tt.z];
  }
  scal[idx] = L[0];
#pragma unroll
  for (int i = 0; i < IRR; ++i) Vi[base + i] = V[i] + L[i];
}

// ---------------------------------------------------------------------------
// Concatenate h = [xi(128) | density[:,:,0](64) | scal0(16) | scal1(16)]
// ---------------------------------------------------------------------------
__global__ void concat_kernel(const float* __restrict__ xi, const float* __restrict__ density,
                              const float* __restrict__ s0, const float* __restrict__ s1,
                              float* __restrict__ h, int Nn) {
  int idx = blockIdx.x * blockDim.x + threadIdx.x;
  if (idx >= Nn * LATIN) return;
  int n = idx / LATIN, j = idx - n * LATIN;
  float v;
  if (j < 128)      v = xi[n * NN_DIM + j];
  else if (j < 192) v = density[((size_t)n * DCH + (j - 128)) * IRR];
  else if (j < 208) v = s0[n * NCH + (j - 192)];
  else              v = s1[n * NCH + (j - 208)];
  h[idx] = v;
}

// ---------------------------------------------------------------------------
// Output: d_out = [xi flat | Vi flat]
// ---------------------------------------------------------------------------
__global__ void write_out_kernel(const float* __restrict__ xi, const float* __restrict__ Vi,
                                 float* __restrict__ out, int Nn) {
  int idx = blockIdx.x * blockDim.x + threadIdx.x;
  int nxi = Nn * NN_DIM;
  int total = nxi + Nn * NCH * IRR;
  if (idx >= total) return;
  out[idx] = (idx < nxi) ? xi[idx] : Vi[idx - nxi];
}

// ---------------------------------------------------------------------------
extern "C" void kernel_launch(void* const* d_in, const int* in_sizes, int n_in,
                              void* d_out, int out_size, void* d_ws, size_t ws_size,
                              hipStream_t stream) {
  const int* species = (const int*)d_in[0];
  const int* esrc    = (const int*)d_in[1];
  const int* edst    = (const int*)d_in[2];
  const float* vec   = (const float*)d_in[3];
  const float* dist  = (const float*)d_in[4];
  const float* sw    = (const float*)d_in[5];
  const float* W_spe  = (const float*)d_in[6];
  const float* b_spe  = (const float*)d_in[7];
  const float* W_Vi   = (const float*)d_in[8];
  const float* W_rho1 = (const float*)d_in[9];
  const float* Wm[2]  = {(const float*)d_in[10], (const float*)d_in[20]};
  const float* bm[2]  = {(const float*)d_in[11], (const float*)d_in[21]};
  const float* WH[2][2]  = {{(const float*)d_in[12], (const float*)d_in[14]},
                            {(const float*)d_in[22], (const float*)d_in[24]}};
  const float* wtp[2][2] = {{(const float*)d_in[13], (const float*)d_in[15]},
                            {(const float*)d_in[23], (const float*)d_in[25]}};
  const float* Wl1[2] = {(const float*)d_in[16], (const float*)d_in[26]};
  const float* bl1[2] = {(const float*)d_in[17], (const float*)d_in[27]};
  const float* Wl2[2] = {(const float*)d_in[18], (const float*)d_in[28]};
  const float* bl2[2] = {(const float*)d_in[19], (const float*)d_in[29]};

  int Nn = in_sizes[0];
  int E  = in_sizes[1];

  // bump allocator on workspace
  char* ws = (char*)d_ws;
  size_t off = 0;
  auto alloc = [&](size_t bytes) -> void* {
    void* p = (void*)(ws + off);
    off += (bytes + 255) & ~(size_t)255;
    return p;
  };
  float* rb      = (float*)alloc((size_t)E * NB * 4);
  float* Ysh     = (float*)alloc((size_t)E * IRR * 4);
  int*   roff    = (int*)alloc((size_t)(Nn + 1) * 4);
  float* xi      = (float*)alloc((size_t)Nn * NN_DIM * 4);
  float* mi      = (float*)alloc((size_t)Nn * MSGC * 4);
  float* density = (float*)alloc((size_t)Nn * DCH * IRR * 4);
  float* Vi      = (float*)alloc((size_t)Nn * NCH * IRR * 4);
  float* rhoi    = (float*)alloc((size_t)Nn * DCH * IRR * 4);
  float* Hi      = (float*)alloc((size_t)Nn * NCH * IRR * 4);
  float* scal0   = (float*)alloc((size_t)Nn * NCH * 4);
  float* scal1   = (float*)alloc((size_t)Nn * NCH * 4);
  float* hbuf    = (float*)alloc((size_t)Nn * LATIN * 4);
  float* hidden  = (float*)alloc((size_t)Nn * NN_DIM * 4);
  int*   tp_cnt  = (int*)alloc(16);
  int4*  tp_idx  = (int4*)alloc((size_t)MAXTP * sizeof(int4));
  float* tp_cf   = (float*)alloc((size_t)MAXTP * 4);
  // f16 staging
  _Float16* A16 = (_Float16*)alloc((size_t)Nn * LATIN * 2);     // max K = 224
  _Float16* WtM[2], *WtL1[2], *WtL2[2];
  for (int l = 0; l < 2; ++l) {
    WtM[l]  = (_Float16*)alloc((size_t)16 * NN_DIM * 2);        // Wm: 128x8 -> 16x128
    WtL1[l] = (_Float16*)alloc((size_t)NN_DIM * LATIN * 2);     // 224x128 -> 128x224
    WtL2[l] = (_Float16*)alloc((size_t)NN_DIM * NN_DIM * 2);    // 128x128 -> 128x128
  }

  auto pack_w = [&](const float* W, _Float16* Wt, int K, int N, int N16) {
    int total = N16 * K;
    pack_w_kernel<<<(total + 255) / 256, 256, 0, stream>>>(W, Wt, K, N, N16);
  };
  auto launch_gemm = [&](const float* A, const _Float16* Wt, const float* B, float* O,
                         int M, int K, int N, int act, int accum) {
    int totalA = M * K;
    pack_a_kernel<<<(totalA + 255) / 256, 256, 0, stream>>>(A, A16, totalA);
    int Mt = (M + 15) / 16, Nt = (N + 15) / 16;
    int tiles = Mt * Nt;
    int blocks = (tiles + 7) / 8;  // 8 waves (256 threads) per block
    gemm_wmma_kernel<<<blocks, 256, 0, stream>>>(A16, Wt, B, O, M, K, N, Nt, tiles,
                                                 act, accum);
  };

  // one-time per-launch prep
  tp_init_kernel<<<1, 32, 0, stream>>>(tp_cnt, tp_idx, tp_cf);
  for (int l = 0; l < 2; ++l) {
    pack_w(Wm[l],  WtM[l],  NN_DIM, MSGC,  16);
    pack_w(Wl1[l], WtL1[l], LATIN,  NN_DIM, NN_DIM);
    pack_w(Wl2[l], WtL2[l], NN_DIM, NN_DIM, NN_DIM);
  }
  edge_feat_kernel<<<(E + 255) / 256, 256, 0, stream>>>(vec, dist, sw, rb, Ysh, E);
  row_offsets_kernel<<<(Nn + 256) / 256, 256, 0, stream>>>(esrc, roff, Nn, E);
  embed_kernel<<<(Nn * NN_DIM + 255) / 256, 256, 0, stream>>>(species, W_spe, b_spe, xi, Nn);

  for (int l = 0; l < 2; ++l) {
    // mi = xi @ Wm + bm   (N=8 padded to one 16-wide tile)
    launch_gemm(xi, WtM[l], bm[l], mi, Nn, NN_DIM, MSGC, 0, 0);

    if (l == 0) {
      density0_kernel<<<Nn, 256, 0, stream>>>(roff, edst, mi, rb, Ysh, density);
      mix_kernel<<<(Nn * NCH * IRR + 255) / 256, 256, 0, stream>>>(
          density, W_Vi, Vi, Nn, DCH, NCH, 1);
    } else {
      mix_kernel<<<(Nn * DCH * IRR + 255) / 256, 256, 0, stream>>>(
          Vi, W_rho1, rhoi, Nn, NCH, DCH, 1);
      density1_kernel<<<Nn, 256, 0, stream>>>(roff, edst, mi, rb, rhoi, density);
    }

    for (int i = 0; i < 2; ++i) {
      mix_kernel<<<(Nn * NCH * IRR + 255) / 256, 256, 0, stream>>>(
          density, WH[l][i], Hi, Nn, DCH, NCH, 0);
      tp_kernel<<<(Nn * NCH + 255) / 256, 256, 0, stream>>>(
          Hi, wtp[l][i], tp_cnt, tp_idx, tp_cf, Vi, (i == 0) ? scal0 : scal1, Nn);
    }

    concat_kernel<<<(Nn * LATIN + 255) / 256, 256, 0, stream>>>(
        xi, density, scal0, scal1, hbuf, Nn);
    launch_gemm(hbuf, WtL1[l], bl1[l], hidden, Nn, LATIN, NN_DIM, 1, 0);   // SiLU
    launch_gemm(hidden, WtL2[l], bl2[l], xi, Nn, NN_DIM, NN_DIM, 0, 1);    // xi += dxi
  }

  write_out_kernel<<<(Nn * (NN_DIM + NCH * IRR) + 255) / 256, 256, 0, stream>>>(
      xi, Vi, (float*)d_out, Nn);

  (void)n_in; (void)out_size; (void)ws_size;
}
